// E_GCL_68539088109876
// MI455X (gfx1250) — matrix operand order repaired and checked by reading, not compiled, not run on hardware
//
#include <hip/hip_runtime.h>

#define NN 50000
#define EE 1000000

typedef __bf16 bf16_t;
typedef __attribute__((ext_vector_type(16))) bf16_t v16bf;
typedef __attribute__((ext_vector_type(8)))  float  v8f;
typedef __attribute__((ext_vector_type(4)))  unsigned int u32x4;
typedef __attribute__((ext_vector_type(4)))  float  f32x4;

union Frag16 { v16bf v; unsigned u[8]; u32x4 q[2]; };
union FragC  { v8f v; float f[8]; };

// ---- f32 -> bf16 conversion -------------------------------------------------
// Preferred: hardware cvt_pk (if toolchain declares it). Fallback: round-half-up
// (u + 0x8000) then a single v_perm_b32 to pack the two high halves.
#if __has_builtin(__builtin_amdgcn_cvt_pk_bf16_f32)
__device__ __forceinline__ unsigned pack_bf16(float a, float b) {
  auto p = __builtin_amdgcn_cvt_pk_bf16_f32(a, b);
  unsigned r; __builtin_memcpy(&r, &p, 4);
  return r;
}
__device__ __forceinline__ unsigned short bf16_1(float a) {
  return (unsigned short)(pack_bf16(a, 0.0f) & 0xFFFFu);
}
#else
__device__ __forceinline__ unsigned pack_bf16(float a, float b) {
  union { float f; unsigned u; } x, y; x.f = a; y.f = b;
  // D = { (b+0x8000)[31:16], (a+0x8000)[31:16] } via one v_perm_b32
  return __builtin_amdgcn_perm(y.u + 0x8000u, x.u + 0x8000u, 0x07060302u);
}
__device__ __forceinline__ unsigned short bf16_1(float a) {
  union { float f; unsigned u; } x; x.f = a;
  return (unsigned short)((x.u + 0x8000u) >> 16);
}
#endif

__device__ __forceinline__ v8f wmma_bf16(const Frag16& A, const Frag16& B, v8f C) {
  return __builtin_amdgcn_wmma_f32_16x16x32_bf16(false, A.v, false, B.v, (short)0, C, false, false);
}

// ---- fragment loaders -------------------------------------------------------
// A fragment (16x32 bf16) from a global f32 row: lane m=lane&15, khi=(lane>>4)*8
// VGPR0-3 <- K = khi..khi+7 ; VGPR4-7 <- K = 16+khi..16+khi+7
__device__ __forceinline__ void load_afrag_g(const float* __restrict__ src, int lane, Frag16& f) {
  int khi = (lane >> 4) << 3;
  f32x4 a0 = *reinterpret_cast<const f32x4*>(src + khi);
  f32x4 a1 = *reinterpret_cast<const f32x4*>(src + khi + 4);
  f32x4 b0 = *reinterpret_cast<const f32x4*>(src + 16 + khi);
  f32x4 b1 = *reinterpret_cast<const f32x4*>(src + 16 + khi + 4);
  f.u[0] = pack_bf16(a0.x, a0.y); f.u[1] = pack_bf16(a0.z, a0.w);
  f.u[2] = pack_bf16(a1.x, a1.y); f.u[3] = pack_bf16(a1.z, a1.w);
  f.u[4] = pack_bf16(b0.x, b0.y); f.u[5] = pack_bf16(b0.z, b0.w);
  f.u[6] = pack_bf16(b1.x, b1.y); f.u[7] = pack_bf16(b1.z, b1.w);
}
// A fragment from LDS staging (16 rows x 64 bf16, padded stride SST)
#define SST 72
__device__ __forceinline__ void load_afrag_lds(const unsigned short* S, int kt, int lane, Frag16& f) {
  int m = lane & 15;
  int khi = (lane >> 4) << 3;
  f.q[0] = *reinterpret_cast<const u32x4*>(S + m*SST + kt*32 + khi);
  f.q[1] = *reinterpret_cast<const u32x4*>(S + m*SST + kt*32 + 16 + khi);
}
// B fragment (32x16 bf16) from LDS transposed weights [n][K], padded stride
__device__ __forceinline__ void load_bfrag(const unsigned short* Wt, int stride, int nt, int kt, int lane, Frag16& f) {
  int n  = nt*16 + (lane & 15);
  int kb = kt*32 + ((lane >> 4) << 3);
  const u32x4* p = reinterpret_cast<const u32x4*>(Wt + n*stride + kb);
  f.q[0] = p[0]; f.q[1] = p[2];
}

// ---- utility kernels --------------------------------------------------------
__global__ void egcl_zero(float* p, int n) {
  int i = blockIdx.x * 256 + threadIdx.x;
  if (i < n) p[i] = 0.0f;
}

__global__ void egcl_prep(const float* __restrict__ We1, const float* __restrict__ We2,
                          const float* __restrict__ Wn1, const float* __restrict__ Wn2,
                          const float* __restrict__ Wc1,
                          unsigned short* We1T, unsigned short* Wn1T,
                          unsigned short* We2T, unsigned short* Wn2T,
                          unsigned short* Wc1T, float* W1last) {
  int i = blockIdx.x * 256 + threadIdx.x;
  if (i < 8192) { int n = i >> 7, k = i & 127;
    We1T[i] = bf16_1(We1[k*64 + n]);
    Wn1T[i] = bf16_1(Wn1[k*64 + n]);
  }
  if (i < 4096) { int n = i >> 6, k = i & 63;
    We2T[i] = bf16_1(We2[k*64 + n]);
    Wn2T[i] = bf16_1(Wn2[k*64 + n]);
    Wc1T[i] = bf16_1(Wc1[k*64 + n]);
  }
  if (i < 64) W1last[i] = We1[128*64 + i];
}

// ---- edge kernel: 16 edges per wave, 8 waves per block, persistent ----------
#define WS1 136   // LDS stride for K=128 weights (bank-conflict-free)
#define WS2 72    // LDS stride for K=64 weights
#define NTILE ((EE + 127) / 128)

__global__ __launch_bounds__(256) void egcl_edge(
    const float* __restrict__ h, const float* __restrict__ coord,
    const int* __restrict__ eidx,
    const float* __restrict__ be1, const float* __restrict__ be2,
    const float* __restrict__ bc1, const float* __restrict__ Wc2,
    const unsigned short* __restrict__ gWe1T, const unsigned short* __restrict__ gWe2T,
    const unsigned short* __restrict__ gWc1T, const float* __restrict__ gW1last,
    float* __restrict__ agg, float* __restrict__ trans_sum, float* __restrict__ cnt)
{
  __shared__ __align__(16) unsigned short We1Ts[64*WS1];
  __shared__ __align__(16) unsigned short We2Ts[64*WS2];
  __shared__ __align__(16) unsigned short Wc1Ts[64*WS2];
  __shared__ __align__(16) unsigned short stage[8][16*SST];
  __shared__ float be1s[64], be2s[64], bc1s[64], W1l[64], Wc2s[64];
  __shared__ float radial_s[8][16], cdx[8][16], cdy[8][16], cdz[8][16], gate_s[8][16];
  __shared__ int rs[8][16], cs[8][16];

  int tid = threadIdx.x;
  for (int i = tid; i < 8192; i += 256) { int n = i >> 7, k = i & 127; We1Ts[n*WS1 + k] = gWe1T[i]; }
  for (int i = tid; i < 4096; i += 256) { int n = i >> 6, k = i & 63;
    We2Ts[n*WS2 + k] = gWe2T[i]; Wc1Ts[n*WS2 + k] = gWc1T[i]; }
  if (tid < 64) { be1s[tid] = be1[tid]; be2s[tid] = be2[tid]; bc1s[tid] = bc1[tid];
                  W1l[tid] = gW1last[tid]; Wc2s[tid] = Wc2[tid]; }
  __syncthreads();

  int wave = tid >> 5, lane = tid & 31;
  int m16 = lane & 15;
  int mh  = (lane >> 4) << 3;

  for (int tb = blockIdx.x; tb < NTILE; tb += gridDim.x) {
    int e0 = (tb * 8 + wave) * 16;
    if (e0 >= EE) continue;   // uniform per wave (E % 16 == 0)

    // Phase A: edge metadata (lanes 0-15, one edge each)
    if (lane < 16) {
      int e = e0 + lane;
      int r = eidx[e], c = eidx[EE + e];
      float dx = coord[r*3+0] - coord[c*3+0];
      float dy = coord[r*3+1] - coord[c*3+1];
      float dz = coord[r*3+2] - coord[c*3+2];
      radial_s[wave][lane] = dx*dx + dy*dy + dz*dz;
      cdx[wave][lane] = dx; cdy[wave][lane] = dy; cdz[wave][lane] = dz;
      rs[wave][lane] = r; cs[wave][lane] = c;
      atomicAdd(&cnt[r], 1.0f);
    }

    int rowm = rs[wave][m16], colm = cs[wave][m16];
    int rowv[8];
    float rad[8];
#pragma unroll
    for (int r = 0; r < 8; ++r) { rowv[r] = rs[wave][mh + r]; rad[r] = radial_s[wave][mh + r]; }

    // Layer 1: [h_i | h_j | radial] @ We1 + be1 ; radial column folded into init
    Frag16 A[4];
    load_afrag_g(h + (size_t)rowm*64 +  0, lane, A[0]);
    load_afrag_g(h + (size_t)rowm*64 + 32, lane, A[1]);
    load_afrag_g(h + (size_t)colm*64 +  0, lane, A[2]);
    load_afrag_g(h + (size_t)colm*64 + 32, lane, A[3]);

    FragC acc[4];
#pragma unroll
    for (int nt = 0; nt < 4; ++nt) {
      int n = nt*16 + m16;
      float b = be1s[n], wl = W1l[n];
#pragma unroll
      for (int r = 0; r < 8; ++r) acc[nt].f[r] = b + rad[r] * wl;
    }
#pragma unroll
    for (int nt = 0; nt < 4; ++nt)
#pragma unroll
      for (int kt = 0; kt < 4; ++kt) {
        Frag16 B; load_bfrag(We1Ts, WS1, nt, kt, lane, B);
        acc[nt].v = wmma_bf16(A[kt], B, acc[nt].v);
      }

    // ReLU -> stage as bf16 A-layout source
    unsigned short* S = stage[wave];
#pragma unroll
    for (int nt = 0; nt < 4; ++nt) {
      int n = nt*16 + m16;
#pragma unroll
      for (int r = 0; r < 8; ++r) S[(mh + r)*SST + n] = bf16_1(fmaxf(acc[nt].f[r], 0.0f));
    }

    // Layer 2: m @ We2 + be2, ReLU
    Frag16 A2[2];
    load_afrag_lds(S, 0, lane, A2[0]);
    load_afrag_lds(S, 1, lane, A2[1]);
#pragma unroll
    for (int nt = 0; nt < 4; ++nt) {
      float b = be2s[nt*16 + m16];
#pragma unroll
      for (int r = 0; r < 8; ++r) acc[nt].f[r] = b;
#pragma unroll
      for (int kt = 0; kt < 2; ++kt) {
        Frag16 B; load_bfrag(We2Ts, WS2, nt, kt, lane, B);
        acc[nt].v = wmma_bf16(A2[kt], B, acc[nt].v);
      }
    }

    // ReLU -> edge_feat: scatter to agg (f32 atomics) + restage for coord gate
#pragma unroll
    for (int r = 0; r < 8; ++r) {
      size_t rowbase = (size_t)rowv[r] * 64;
#pragma unroll
      for (int nt = 0; nt < 4; ++nt) {
        int n = nt*16 + m16;
        float v = fmaxf(acc[nt].f[r], 0.0f);
        atomicAdd(&agg[rowbase + n], v);
        S[(mh + r)*SST + n] = bf16_1(v);
      }
    }

    // Coord gate: relu(edge_feat @ Wc1 + bc1) @ Wc2
    load_afrag_lds(S, 0, lane, A2[0]);
    load_afrag_lds(S, 1, lane, A2[1]);
    FragC g[4];
#pragma unroll
    for (int nt = 0; nt < 4; ++nt) {
      float b = bc1s[nt*16 + m16];
#pragma unroll
      for (int r = 0; r < 8; ++r) g[nt].f[r] = b;
#pragma unroll
      for (int kt = 0; kt < 2; ++kt) {
        Frag16 B; load_bfrag(Wc1Ts, WS2, nt, kt, lane, B);
        g[nt].v = wmma_bf16(A2[kt], B, g[nt].v);
      }
    }
    // reduce 64 -> 1 per edge: partial per lane, xor-shuffle across the 16-lane half
#pragma unroll
    for (int r = 0; r < 8; ++r) {
      float s = 0.0f;
#pragma unroll
      for (int nt = 0; nt < 4; ++nt) s += fmaxf(g[nt].f[r], 0.0f) * Wc2s[nt*16 + m16];
      s += __shfl_xor(s, 1, 32);
      s += __shfl_xor(s, 2, 32);
      s += __shfl_xor(s, 4, 32);
      s += __shfl_xor(s, 8, 32);
      if (m16 == 0) gate_s[wave][mh + r] = s;
    }

    if (lane < 16) {
      float gg = gate_s[wave][lane];
      float tx = fminf(fmaxf(cdx[wave][lane]*gg, -100.0f), 100.0f);
      float ty = fminf(fmaxf(cdy[wave][lane]*gg, -100.0f), 100.0f);
      float tz = fminf(fmaxf(cdz[wave][lane]*gg, -100.0f), 100.0f);
      int r = rs[wave][lane];
      atomicAdd(&trans_sum[r*3 + 0], tx);
      atomicAdd(&trans_sum[r*3 + 1], ty);
      atomicAdd(&trans_sum[r*3 + 2], tz);
    }
  }
}

// ---- node kernel: h_out = h + relu([h|agg]@Wn1+bn1)@Wn2 + bn2 ---------------
__global__ __launch_bounds__(256) void egcl_node(
    const float* __restrict__ h, const float* __restrict__ agg,
    const float* __restrict__ bn1, const float* __restrict__ bn2,
    const unsigned short* __restrict__ gWn1T, const unsigned short* __restrict__ gWn2T,
    float* __restrict__ hout)
{
  __shared__ __align__(16) unsigned short Wn1Ts[64*WS1];
  __shared__ __align__(16) unsigned short Wn2Ts[64*WS2];
  __shared__ __align__(16) unsigned short stage[8][16*SST];
  __shared__ float bn1s[64], bn2s[64];

  int tid = threadIdx.x;
  for (int i = tid; i < 8192; i += 256) { int n = i >> 7, k = i & 127; Wn1Ts[n*WS1 + k] = gWn1T[i]; }
  for (int i = tid; i < 4096; i += 256) { int n = i >> 6, k = i & 63;  Wn2Ts[n*WS2 + k] = gWn2T[i]; }
  if (tid < 64) { bn1s[tid] = bn1[tid]; bn2s[tid] = bn2[tid]; }
  __syncthreads();

  int wave = tid >> 5, lane = tid & 31;
  int i0 = (blockIdx.x * 8 + wave) * 16;
  if (i0 >= NN) return;   // whole wave uniform (N % 16 == 0)

  int m16 = lane & 15;
  int mh  = (lane >> 4) << 3;
  int node = i0 + m16;

  Frag16 A[4];
  load_afrag_g(h   + (size_t)node*64 +  0, lane, A[0]);
  load_afrag_g(h   + (size_t)node*64 + 32, lane, A[1]);
  load_afrag_g(agg + (size_t)node*64 +  0, lane, A[2]);
  load_afrag_g(agg + (size_t)node*64 + 32, lane, A[3]);

  FragC acc[4];
#pragma unroll
  for (int nt = 0; nt < 4; ++nt) {
    float b = bn1s[nt*16 + m16];
#pragma unroll
    for (int r = 0; r < 8; ++r) acc[nt].f[r] = b;
#pragma unroll
    for (int kt = 0; kt < 4; ++kt) {
      Frag16 B; load_bfrag(Wn1Ts, WS1, nt, kt, lane, B);
      acc[nt].v = wmma_bf16(A[kt], B, acc[nt].v);
    }
  }

  unsigned short* S = stage[wave];
#pragma unroll
  for (int nt = 0; nt < 4; ++nt) {
    int n = nt*16 + m16;
#pragma unroll
    for (int r = 0; r < 8; ++r) S[(mh + r)*SST + n] = bf16_1(fmaxf(acc[nt].f[r], 0.0f));
  }

  Frag16 A2[2];
  load_afrag_lds(S, 0, lane, A2[0]);
  load_afrag_lds(S, 1, lane, A2[1]);
#pragma unroll
  for (int nt = 0; nt < 4; ++nt) {
    float b = bn2s[nt*16 + m16];
#pragma unroll
    for (int r = 0; r < 8; ++r) acc[nt].f[r] = b;
#pragma unroll
    for (int kt = 0; kt < 2; ++kt) {
      Frag16 B; load_bfrag(Wn2Ts, WS2, nt, kt, lane, B);
      acc[nt].v = wmma_bf16(A2[kt], B, acc[nt].v);
    }
  }

  // residual add + store
#pragma unroll
  for (int nt = 0; nt < 4; ++nt) {
    int n = nt*16 + m16;
#pragma unroll
    for (int r = 0; r < 8; ++r) {
      size_t idx = (size_t)(i0 + mh + r)*64 + n;
      hout[idx] = h[idx] + acc[nt].f[r];
    }
  }
}

__global__ void egcl_coordout(const float* __restrict__ trans_sum,
                              const float* __restrict__ cnt, float* __restrict__ out) {
  int i = blockIdx.x * 256 + threadIdx.x;
  if (i >= NN*3) return;
  int node = i / 3;
  out[(size_t)NN*64 + i] = trans_sum[i] / fmaxf(cnt[node], 1.0f);
}

// ---- launch -----------------------------------------------------------------
extern "C" void kernel_launch(void* const* d_in, const int* in_sizes, int n_in,
                              void* d_out, int out_size, void* d_ws, size_t ws_size,
                              hipStream_t stream) {
  const float* h     = (const float*)d_in[0];
  const float* coord = (const float*)d_in[1];
  const int*   eidx  = (const int*)  d_in[2];
  const float* We1   = (const float*)d_in[3];
  const float* be1   = (const float*)d_in[4];
  const float* We2   = (const float*)d_in[5];
  const float* be2   = (const float*)d_in[6];
  const float* Wn1   = (const float*)d_in[7];
  const float* bn1   = (const float*)d_in[8];
  const float* Wn2   = (const float*)d_in[9];
  const float* bn2   = (const float*)d_in[10];
  const float* Wc1   = (const float*)d_in[11];
  const float* bc1   = (const float*)d_in[12];
  const float* Wc2   = (const float*)d_in[13];

  // workspace layout
  float* agg   = (float*)d_ws;                 // N*64
  float* trans = agg + (size_t)NN*64;          // N*3
  float* cnt   = trans + (size_t)NN*3;         // N
  unsigned short* We1T = (unsigned short*)(cnt + NN);   // 8192
  unsigned short* Wn1T = We1T + 8192;                   // 8192
  unsigned short* We2T = Wn1T + 8192;                   // 4096
  unsigned short* Wn2T = We2T + 4096;                   // 4096
  unsigned short* Wc1T = Wn2T + 4096;                   // 4096
  float* W1last = (float*)(Wc1T + 4096);                // 64

  float* hout = (float*)d_out;

  // 1) zero accumulators (agg | trans | cnt are contiguous: N*68 floats)
  {
    int n = NN * 68;
    egcl_zero<<<(n + 255)/256, 256, 0, stream>>>(agg, n);
  }
  // 2) transpose + bf16-convert weights
  egcl_prep<<<32, 256, 0, stream>>>(We1, We2, Wn1, Wn2, Wc1,
                                    We1T, Wn1T, We2T, Wn2T, Wc1T, W1last);
  // 3) edge phase: persistent blocks, grid-stride over 128-edge tiles
  {
    int grid = NTILE < 2048 ? NTILE : 2048;
    egcl_edge<<<grid, 256, 0, stream>>>(h, coord, eidx, be1, be2, bc1, Wc2,
                                        We1T, We2T, Wc1T, W1last,
                                        agg, trans, cnt);
  }
  // 4) node phase: 128 nodes/block
  egcl_node<<<(NN + 127)/128, 256, 0, stream>>>(h, agg, bn1, bn2, Wn1T, Wn2T, hout);
  // 5) coord output
  egcl_coordout<<<(NN*3 + 255)/256, 256, 0, stream>>>(trans, cnt, hout);
}
